// qNetwork_SNN_dynamic_76733885710365
// MI455X (gfx1250) — matrix operand order repaired and checked by reading, not compiled, not run on hardware
//
#include <hip/hip_runtime.h>

// ---------------------------------------------------------------------------
// SNN forward (3x Linear+Leaky, T=32) for MI455X (gfx1250, wave32).
// Single persistent kernel, 96 dependent layer-steps with a device-scope
// grid barrier. GEMMs on v_wmma_f32_16x16x32_bf16 (fp32 accum), LIF fused.
//
// Data-movement design (the actual bottleneck):
//   * weights bf16 (6MB) + state (8MB) are L2-resident (192MB) -> HBM ~0.
//   * naive per-wave streaming would pull ~84MB/step from L2 (~8GB total).
//   * instead each 256-thread workgroup (8 waves, 128 M-rows) stages its
//     64-col weight strip (64x1024 bf16 = 128KB, padded) in LDS ONCE per
//     step and runs all 32 K-steps from LDS: 24MB/step (~2.3GB), 3.5x less.
// ---------------------------------------------------------------------------

typedef __attribute__((ext_vector_type(16))) __bf16 v16bf;
typedef __attribute__((ext_vector_type(8)))  float  v8f;

#define BROWS   512
#define DIM     1024
#define NSTEPS  32
#define NLAYER  3
#define NBLOCKS 64          // 4 M-groups (128 rows) x 16 N-strips (64 cols)
#define TPB     256         // 8 waves
#define LDSPAD  4           // +4 dwords/row: bank spread, keeps 16B align
#define LROW    (DIM / 2 + LDSPAD)   // 516 dwords per LDS weight row

union FragBF { v16bf v; unsigned int u[8]; };

__device__ __forceinline__ unsigned short f32_to_bf16(float f) {
  unsigned int u = __float_as_uint(f);
  u += 0x7FFFu + ((u >> 16) & 1u);        // round-to-nearest-even
  return (unsigned short)(u >> 16);
}

// ---------------------------------------------------------------------------
// Init: W* fp32 -> bf16, x fp32 -> bf16, zero membranes/accumulator/barrier.
// ---------------------------------------------------------------------------
__global__ void snn_init_kernel(const float* __restrict__ x,
                                const float* __restrict__ W0,
                                const float* __restrict__ W1,
                                const float* __restrict__ W2,
                                unsigned short* __restrict__ Wbf,
                                unsigned short* __restrict__ xbf,
                                float* __restrict__ mem,
                                float* __restrict__ acc,
                                unsigned int* __restrict__ barrier_cnt) {
  const int NW = NLAYER * DIM * DIM;                       // 3,145,728
  for (int i = blockIdx.x * blockDim.x + threadIdx.x; i < NW;
       i += gridDim.x * blockDim.x) {
    const int l = i >> 20;                                 // / (1024*1024)
    const int r = i & 0xFFFFF;
    const float w = (l == 0) ? W0[r] : (l == 1) ? W1[r] : W2[r];
    Wbf[i] = f32_to_bf16(w);
    if (i < NLAYER * BROWS * DIM) mem[i] = 0.0f;
    if (i < BROWS * DIM) { acc[i] = 0.0f; xbf[i] = f32_to_bf16(x[i]); }
    if (i == 0) *barrier_cnt = 0u;
  }
}

// ---------------------------------------------------------------------------
// Persistent kernel.
//   block b: mgroup = b>>4 (128 rows), nstrip = b&15 (64 cols)
//   wave w (0..7): mtile = mgroup*8 + w  -> 16 output rows
//   Per step: stage weight strip in LDS, 32 K-steps of 4 WMMAs, LIF epilogue,
//   then device-scope grid barrier (monotonic target, s_sleep spin).
// ---------------------------------------------------------------------------
__global__ void __launch_bounds__(TPB)
snn_persistent_kernel(const unsigned int* __restrict__ Wbf_u,  // 3x[1024,1024] bf16
                      unsigned short*     __restrict__ xb0,
                      unsigned short*     __restrict__ xb1,
                      const float* __restrict__ b0,
                      const float* __restrict__ b1,
                      const float* __restrict__ b2,
                      float* __restrict__ mem_all,             // 3x[512,1024]
                      float* __restrict__ acc,                 // d_out
                      unsigned int* __restrict__ barrier_cnt)
{
  __shared__ unsigned int lB[64][LROW];   // 64 weight rows (cols of W^T), 132KB

  const int lane   = threadIdx.x & 31;
  const int wave   = threadIdx.x >> 5;                 // 0..7
  const int mgroup = blockIdx.x >> 4;                  // 0..3
  const int nstrip = blockIdx.x & 15;                  // 0..15
  const int mtile  = mgroup * 8 + wave;                // 0..31
  const int n      = lane & 15;
  const int hf     = lane >> 4;
  const int obase  = nstrip * 64;
  const int HSTRIDE = DIM / 2;                         // global row stride (dwords)
  const int arow_m  = mtile * 16 + n;                  // A-matrix row for lane

  for (int s = 0; s < NSTEPS * NLAYER; ++s) {
    const int layer = s - (s / NLAYER) * NLAYER;
    const unsigned int* w_u = Wbf_u + (size_t)layer * DIM * HSTRIDE;
    const float* bias = (layer == 0) ? b0 : (layer == 1) ? b1 : b2;
    float* mem = mem_all + (size_t)layer * BROWS * DIM;
    const unsigned int* xin_u = (const unsigned int*)((s & 1) ? xb1 : xb0);
    unsigned short* xout = (s & 1) ? xb0 : xb1;

    // ---- stage full 64x1024 bf16 weight strip into LDS (uint4 copies) ----
    // 64 rows x 128 uint4 = 8192 uint4; 256 threads x 32 each.
#pragma unroll
    for (int j = 0; j < 32; ++j) {
      const int i4  = threadIdx.x + TPB * j;           // 0..8191
      const int row = i4 >> 7;                         // 128 uint4 per row
      const int c4  = i4 & 127;
      const uint4 v = *(const uint4*)(w_u + (size_t)(obase + row) * HSTRIDE +
                                      (size_t)c4 * 4);
      *(uint4*)&lB[row][c4 * 4] = v;
    }
    __syncthreads();

    // ---- GEMM: 32 K-steps x 4 N-tiles from LDS, A direct from global ----
    const unsigned int* arow = xin_u + (size_t)arow_m * HSTRIDE;
    v8f c0 = {}, c1 = {}, c2 = {}, c3 = {};

    for (int ks = 0; ks < DIM / 32; ++ks) {
      // A fragment: 16x32 bf16 (contiguous K: k = 16*hf + s, matches B)
      FragBF A;
#pragma unroll
      for (int v = 0; v < 8; ++v) {
        const int uidx = v + 8 * hf;
        A.u[v] = arow[ks * 16 + uidx];
      }
      if (ks + 1 < DIM / 32)
        __builtin_prefetch(arow + (ks + 1) * 16, 0, 3);   // global_prefetch

      // B fragments from LDS: 32x16 bf16 (K over dword pairs, N over lanes)
      FragBF Bf;
#pragma unroll
      for (int v = 0; v < 8; ++v) Bf.u[v] = lB[ 0 + n][ks * 16 + v + 8 * hf];
      c0 = __builtin_amdgcn_wmma_f32_16x16x32_bf16(false, A.v, false, Bf.v,
                                                   (short)0, c0, false, false);
#pragma unroll
      for (int v = 0; v < 8; ++v) Bf.u[v] = lB[16 + n][ks * 16 + v + 8 * hf];
      c1 = __builtin_amdgcn_wmma_f32_16x16x32_bf16(false, A.v, false, Bf.v,
                                                   (short)0, c1, false, false);
#pragma unroll
      for (int v = 0; v < 8; ++v) Bf.u[v] = lB[32 + n][ks * 16 + v + 8 * hf];
      c2 = __builtin_amdgcn_wmma_f32_16x16x32_bf16(false, A.v, false, Bf.v,
                                                   (short)0, c2, false, false);
#pragma unroll
      for (int v = 0; v < 8; ++v) Bf.u[v] = lB[48 + n][ks * 16 + v + 8 * hf];
      c3 = __builtin_amdgcn_wmma_f32_16x16x32_bf16(false, A.v, false, Bf.v,
                                                   (short)0, c3, false, false);
    }

    // ---- fused LIF epilogue (C layout: VGPR r -> M = r + 8*hf, N = n) ----
#pragma unroll
    for (int t = 0; t < 4; ++t) {
      const v8f cc = (t == 0) ? c0 : (t == 1) ? c1 : (t == 2) ? c2 : c3;
      const int col = obase + t * 16 + n;
      const float bv = bias[col];
#pragma unroll
      for (int r = 0; r < 8; ++r) {
        const int row = mtile * 16 + r + 8 * hf;
        const int idx = row * DIM + col;
        const float cur = cc[r] + bv;
        const float m_old = mem[idx];
        const float rst = (m_old > 1.0f) ? 1.0f : 0.0f;   // pre-update reset
        const float m_new = 0.9f * m_old + cur - rst;     // subtract threshold
        const float spk = (m_new > 1.0f) ? 1.0f : 0.0f;
        mem[idx] = m_new;
        acc[idx] = acc[idx] + cur + spk;                  // outSpikes sum
        xout[idx] = (m_new > 1.0f) ? (unsigned short)0x3F80u
                                   : (unsigned short)0u;  // bf16 spike
      }
    }

    // ---- device-scope grid barrier (also covers LDS WAR for next step) ----
    __threadfence();                 // release: flush spikes/mem to DEV scope
    __syncthreads();
    if (threadIdx.x == 0) {
      __hip_atomic_fetch_add(barrier_cnt, 1u, __ATOMIC_RELEASE,
                             __HIP_MEMORY_SCOPE_AGENT);
      const unsigned int target = (unsigned int)(s + 1) * NBLOCKS;
      while (__hip_atomic_load(barrier_cnt, __ATOMIC_ACQUIRE,
                               __HIP_MEMORY_SCOPE_AGENT) < target)
        __builtin_amdgcn_s_sleep(1);
    }
    __syncthreads();
    __threadfence();                 // acquire: invalidate stale near caches
  }
}

// ---------------------------------------------------------------------------
// Workspace layout (bytes):
//   [0,        6291456)  Wbf   : 3 x 1024x1024 bf16
//   [6291456,  7340032)  xbuf0 : 512x1024 bf16
//   [7340032,  8388608)  xbuf1 : 512x1024 bf16
//   [8388608, 14680064)  mem   : 3 x 512x1024 fp32
//   [14680064,14680068)  grid-barrier counter
// ---------------------------------------------------------------------------
extern "C" void kernel_launch(void* const* d_in, const int* in_sizes, int n_in,
                              void* d_out, int out_size, void* d_ws, size_t ws_size,
                              hipStream_t stream) {
  (void)in_sizes; (void)n_in; (void)out_size; (void)ws_size;
  const float* x  = (const float*)d_in[0];
  const float* W0 = (const float*)d_in[1];
  const float* b0 = (const float*)d_in[2];
  const float* W1 = (const float*)d_in[3];
  const float* b1 = (const float*)d_in[4];
  const float* W2 = (const float*)d_in[5];
  const float* b2 = (const float*)d_in[6];
  float* acc = (float*)d_out;

  char* ws = (char*)d_ws;
  unsigned short* Wbf = (unsigned short*)(ws);
  unsigned short* xb0 = (unsigned short*)(ws + 6291456u);
  unsigned short* xb1 = (unsigned short*)(ws + 7340032u);
  float*          mem = (float*)(ws + 8388608u);
  unsigned int*   cnt = (unsigned int*)(ws + 14680064u);

  snn_init_kernel<<<1536, 256, 0, stream>>>(x, W0, W1, W2, Wbf, xb0, mem, acc,
                                            cnt);

  snn_persistent_kernel<<<NBLOCKS, TPB, 0, stream>>>(
      (const unsigned int*)Wbf, xb0, xb1, b0, b1, b2, mem, acc, cnt);
}
